// AstPathEncoder_45930380264176
// MI455X (gfx1250) — matrix operand order repaired and checked by reading, not compiled, not run on hardware
//
#include <hip/hip_runtime.h>
#include <hip/hip_bf16.h>
#include <math.h>

typedef __attribute__((ext_vector_type(16))) _Float16 v16h;
typedef __attribute__((ext_vector_type(8)))  _Float16 v8h;
typedef __attribute__((ext_vector_type(8)))  float    v8f;
typedef __attribute__((ext_vector_type(4)))  float    v4f;

// global (addrspace 1) pointer/vector types: force global_load_* instead of
// flat_load_* so weight streams only touch LOADcnt, not DScnt (FLAT bumps both
// and serializes against the LDS A-fragment path on CDNA5).
typedef __attribute__((address_space(1))) const _Float16 GH;
typedef __attribute__((address_space(1))) const float    GF;
typedef __attribute__((address_space(1))) const int      GI;
typedef __attribute__((address_space(1))) const v16h     GV16H;
typedef __attribute__((address_space(1))) const v4f      GV4F;

#define T_LEN 16
#define E_DIM 256
#define H_DIM 256
#define D_DIM 512
#define B_SEG 128

#define LDS_STRIDE 264          // halves; 528B rows: 16B-aligned, 132 dwords -> bank-conflict-free
#define GATE_NTILES 64          // 4H/16 = 1024/16
#define GATE_KTILES 8           // 256/32
#define LIN_NTILES  32          // 512/16
#define LIN_KTILES  16          // 512/32
#define MAT_HALVES  (1024*256)  // halves per packed matrix (Wlin 512x512 == same count)

// ---------------- WMMA helpers ----------------

__device__ __forceinline__ v8f wmma_f16(v16h a, v16h b, v8f c) {
  // D = A(16x32 f16) * B(32x16 f16) + C(16x16 f32)
  return __builtin_amdgcn_wmma_f32_16x16x32_f16(
      false, a, false, b, (short)0, c, false, false);
}

// A-fragment from LDS matrix [16][LDS_STRIDE] halves, K-chunk starting at k0.
// ISA 16-bit A layout: lanes 0-15 row M=lane, elems 0-7 -> K k0..k0+7, 8-15 -> k0+16..k0+23
//                      lanes 16-31 same rows, elems 0-7 -> k0+8..k0+15, 8-15 -> k0+24..k0+31
__device__ __forceinline__ v16h load_a_frag(const _Float16* m, int k0, int lane) {
  const int row = lane & 15;
  const int hi  = lane >> 4;
  const _Float16* p = m + row * LDS_STRIDE + k0 + hi * 8;
  v8h lo = *(const v8h*)(p);
  v8h up = *(const v8h*)(p + 16);
  v16h r;
#pragma unroll
  for (int e = 0; e < 8; ++e) { r[e] = lo[e]; r[e + 8] = up[e]; }
  return r;
}

// B-fragment from fragment-major packed weights in global memory:
// frag = kt*nTiles + nt; each fragment = 32 lanes x 16 contiguous halves
// (one 32B global load per lane).
__device__ __forceinline__ v16h load_b_frag(const GH* base, int kt, int nt,
                                            int nTiles, int lane) {
  return *(const GV16H*)(base + ((size_t)(kt * nTiles + nt)) * 512 + lane * 16);
}

__device__ __forceinline__ float sigmoidf_(float x) {
  return 1.0f / (1.0f + __expf(-x));
}

// ---------------- kernel 0: zero segment sums ----------------

__global__ void zero_sums_kernel(float* s, int n) {
  int i = blockIdx.x * blockDim.x + threadIdx.x;
  if (i < n) s[i] = 0.0f;
}

// ---------------- kernel 1: pack f32 weights -> fragment-major f16 ----------------
// Packed element (mat, frag, lane, e) <- W[nT*16 + (lane&15)][kT*32 + (lane>>4)*8 + (e&7) + (e>>3)*16]

__global__ void pack_weights_kernel(const float* Wih_f, const float* Whh_f,
                                    const float* Wih_b, const float* Whh_b,
                                    const float* Wlin, _Float16* out) {
  int idx = blockIdx.x * blockDim.x + threadIdx.x;
  if (idx >= 5 * MAT_HALVES) return;
  int mat = idx / MAT_HALVES;
  int r   = idx % MAT_HALVES;

  const float* W;
  int nTiles, Kdim;
  if      (mat == 0) { W = Wih_f; nTiles = GATE_NTILES; Kdim = E_DIM; }
  else if (mat == 1) { W = Whh_f; nTiles = GATE_NTILES; Kdim = H_DIM; }
  else if (mat == 2) { W = Wih_b; nTiles = GATE_NTILES; Kdim = E_DIM; }
  else if (mat == 3) { W = Whh_b; nTiles = GATE_NTILES; Kdim = H_DIM; }
  else               { W = Wlin;  nTiles = LIN_NTILES;  Kdim = D_DIM; }

  int e    = r & 15;
  int lane = (r >> 4) & 31;
  int frag = r >> 9;
  int kT   = frag / nTiles;
  int nT   = frag % nTiles;
  int n = nT * 16 + (lane & 15);
  int k = kT * 32 + (lane >> 4) * 8 + (e & 7) + (e >> 3) * 16;
  out[idx] = (_Float16)((const GF*)W)[(size_t)n * Kdim + k];
}

// ---------------- kernel 2: bidirectional LSTM + linear + segment atomic-sum ----------------
// 1 block = 16 sequences, 16 waves (512 threads); wave w owns hidden block [16w,16w+16).

__global__ __launch_bounds__(512, 1)
void lstm_encode_kernel(const int* __restrict__ tok_,
                        const float* __restrict__ emb_,
                        const float* __restrict__ b_f,
                        const float* __restrict__ b_b,
                        const float* __restrict__ b_lin,
                        const _Float16* __restrict__ wp_,
                        float* __restrict__ sums) {
  __shared__ _Float16 xls[16 * LDS_STRIDE];
  __shared__ _Float16 hF [16 * LDS_STRIDE];
  __shared__ _Float16 hB [16 * LDS_STRIDE];
  __shared__ int lenLds[16];

  // force global address space for all streamed operands
  const GI* tok = (const GI*)tok_;
  const GF* emb = (const GF*)emb_;
  const GH* wp  = (const GH*)wp_;

  const int lane = threadIdx.x & 31;
  const int wave = threadIdx.x >> 5;      // 0..15
  const int seqBase = blockIdx.x * 16;
  const int colL = lane & 15;             // column within 16-wide tile
  const int m0 = (lane >> 4) * 8;         // first row (sequence) held by this lane

  const GH* Wih[2] = { wp,              wp + 2 * MAT_HALVES };
  const GH* Whh[2] = { wp + MAT_HALVES, wp + 3 * MAT_HALVES };
  const GH* Wlin   =   wp + 4 * MAT_HALVES;
  const float* bias[2] = { b_f, b_b };

  // per-sequence lengths (pad token == 0)
  if (threadIdx.x < 16) {
    int cnt = 0;
#pragma unroll
    for (int t = 0; t < T_LEN; ++t)
      cnt += (tok[(size_t)(seqBase + threadIdx.x) * T_LEN + t] != 0);
    lenLds[threadIdx.x] = cnt;
  }
  for (int i = threadIdx.x; i < 16 * LDS_STRIDE; i += 512) {
    hF[i] = (_Float16)0.0f;
    hB[i] = (_Float16)0.0f;
  }
  __syncthreads();

  for (int dir = 0; dir < 2; ++dir) {
    _Float16* hC = dir ? hB : hF;
    const GH* Wi = Wih[dir];
    const GH* Wh = Whh[dir];
    const float* bb = bias[dir];

    // per-wave bias values for this hidden block (column = 16*wave + colL)
    const int col = wave * 16 + colL;
    const float bi = bb[0 * H_DIM + col];
    const float bf = bb[1 * H_DIM + col];
    const float bg = bb[2 * H_DIM + col];
    const float bo = bb[3 * H_DIM + col];

    v8f c = {};  // cell state, C/D register layout

    for (int sIdx = 0; sIdx < T_LEN; ++sIdx) {
      const int t = dir ? (T_LEN - 1 - sIdx) : sIdx;

      // gather x_t: wave w fetches embedding row for sequence (seqBase+w)
      {
        const int token = tok[(size_t)(seqBase + wave) * T_LEN + t];
        const GF* er = emb + (size_t)token * E_DIM + lane * 8;
        v4f a0 = *(const GV4F*)(er);
        v4f a1 = *(const GV4F*)(er + 4);
        v8h xv;
        xv[0] = (_Float16)a0.x; xv[1] = (_Float16)a0.y;
        xv[2] = (_Float16)a0.z; xv[3] = (_Float16)a0.w;
        xv[4] = (_Float16)a1.x; xv[5] = (_Float16)a1.y;
        xv[6] = (_Float16)a1.z; xv[7] = (_Float16)a1.w;
        *(v8h*)(xls + wave * LDS_STRIDE + lane * 8) = xv;
      }
      __syncthreads();   // x_t ready; prev-step h writes ready

      v8f ai, af, ag, ao;
#pragma unroll
      for (int r = 0; r < 8; ++r) { ai[r] = bi; af[r] = bf; ag[r] = bg; ao[r] = bo; }

#pragma unroll
      for (int kt = 0; kt < GATE_KTILES; ++kt) {   // x @ W_ih^T
        v16h a = load_a_frag(xls, kt * 32, lane);
        ai = wmma_f16(a, load_b_frag(Wi, kt, wave +  0, GATE_NTILES, lane), ai);
        af = wmma_f16(a, load_b_frag(Wi, kt, wave + 16, GATE_NTILES, lane), af);
        ag = wmma_f16(a, load_b_frag(Wi, kt, wave + 32, GATE_NTILES, lane), ag);
        ao = wmma_f16(a, load_b_frag(Wi, kt, wave + 48, GATE_NTILES, lane), ao);
      }
#pragma unroll
      for (int kt = 0; kt < GATE_KTILES; ++kt) {   // h @ W_hh^T
        v16h a = load_a_frag(hC, kt * 32, lane);
        ai = wmma_f16(a, load_b_frag(Wh, kt, wave +  0, GATE_NTILES, lane), ai);
        af = wmma_f16(a, load_b_frag(Wh, kt, wave + 16, GATE_NTILES, lane), af);
        ag = wmma_f16(a, load_b_frag(Wh, kt, wave + 32, GATE_NTILES, lane), ag);
        ao = wmma_f16(a, load_b_frag(Wh, kt, wave + 48, GATE_NTILES, lane), ao);
      }
      __syncthreads();   // all waves done reading xls / hC

      // elementwise LSTM cell update + packed-sequence length mask
#pragma unroll
      for (int r = 0; r < 8; ++r) {
        const int s = m0 + r;                       // sequence row
        const float iv = sigmoidf_(ai[r]);
        const float fv = sigmoidf_(af[r]);
        const float gv = tanhf(ag[r]);
        const float ov = sigmoidf_(ao[r]);
        const float cn = fv * c[r] + iv * gv;
        const float hn = ov * tanhf(cn);
        const bool msk = (t < lenLds[s]);
        _Float16* hp = hC + s * LDS_STRIDE + col;
        const float hold = (float)(*hp);
        c[r] = msk ? cn : c[r];
        *hp = (_Float16)(msk ? hn : hold);
      }
      // next iteration's gather (xls write) is safe: all waves passed the barrier above
    }
    __syncthreads();
  }

  // ---------------- final linear: repr[16,512] = [hF|hB] @ W_lin^T + b_lin ----------------
  float* segSum = sums + (size_t)(blockIdx.x >> 3) * D_DIM;   // 8 blocks per 128-seq segment

#pragma unroll
  for (int tb = 0; tb < 2; ++tb) {
    const int nT = wave + tb * 16;          // output column tile (0..31)
    const int colg = nT * 16 + colL;
    const float bl = b_lin[colg];
    v8f acc;
#pragma unroll
    for (int r = 0; r < 8; ++r) acc[r] = bl;

#pragma unroll
    for (int kt = 0; kt < LIN_KTILES; ++kt) {
      const _Float16* hsrc = (kt < 8) ? hF : hB;
      v16h a = load_a_frag(hsrc, (kt & 7) * 32, lane);
      v16h b = load_b_frag(Wlin, kt, nT, LIN_NTILES, lane);
      acc = wmma_f16(a, b, acc);
    }
    // pre-reduce this lane's 8 rows (all 16 rows of tile share one segment), one atomic
    float local = 0.0f;
#pragma unroll
    for (int r = 0; r < 8; ++r) local += acc[r];
    __hip_atomic_fetch_add(&segSum[colg], local,
                           __ATOMIC_RELAXED, __HIP_MEMORY_SCOPE_AGENT);
  }
}

// ---------------- kernel 3: mean + L2 normalize ----------------

__global__ __launch_bounds__(256)
void finalize_kernel(const float* __restrict__ sums,
                     const int* __restrict__ seg_len,
                     float* __restrict__ out) {
  __shared__ float red[256];
  const int b = blockIdx.x;
  const float inv = 1.0f / (float)seg_len[b];
  const float v0 = sums[(size_t)b * D_DIM + threadIdx.x] * inv;
  const float v1 = sums[(size_t)b * D_DIM + 256 + threadIdx.x] * inv;
  red[threadIdx.x] = v0 * v0 + v1 * v1;
  __syncthreads();
  for (int off = 128; off > 0; off >>= 1) {
    if (threadIdx.x < off) red[threadIdx.x] += red[threadIdx.x + off];
    __syncthreads();
  }
  const float scale = 1.0f / fmaxf(sqrtf(red[0]), 1e-5f);
  out[(size_t)b * D_DIM + threadIdx.x] = v0 * scale;
  out[(size_t)b * D_DIM + 256 + threadIdx.x] = v1 * scale;
}

// ---------------- launch ----------------

extern "C" void kernel_launch(void* const* d_in, const int* in_sizes, int n_in,
                              void* d_out, int out_size, void* d_ws, size_t ws_size,
                              hipStream_t stream) {
  const int*   tok    = (const int*)  d_in[0];
  const int*   plen   = (const int*)  d_in[1];
  const float* emb    = (const float*)d_in[2];
  const float* Wih_f  = (const float*)d_in[3];
  const float* Whh_f  = (const float*)d_in[4];
  const float* b_f    = (const float*)d_in[5];
  const float* Wih_b  = (const float*)d_in[6];
  const float* Whh_b  = (const float*)d_in[7];
  const float* b_b    = (const float*)d_in[8];
  const float* Wlin   = (const float*)d_in[9];
  const float* b_lin  = (const float*)d_in[10];
  float*       out    = (float*)d_out;

  _Float16* wp   = (_Float16*)d_ws;                              // 5*262144 halves
  float*    sums = (float*)((char*)d_ws + (size_t)5 * MAT_HALVES * 2);

  const int packN = 5 * MAT_HALVES;
  pack_weights_kernel<<<(packN + 255) / 256, 256, 0, stream>>>(
      Wih_f, Whh_f, Wih_b, Whh_b, Wlin, wp);

  zero_sums_kernel<<<(B_SEG * D_DIM + 255) / 256, 256, 0, stream>>>(
      sums, B_SEG * D_DIM);

  lstm_encode_kernel<<<1024, 512, 0, stream>>>(
      tok, emb, b_f, b_b, b_lin, wp, sums);

  finalize_kernel<<<B_SEG, 256, 0, stream>>>(sums, plen, out);
}
